// RadiomicsRetrieval_76527727280647
// MI455X (gfx1250) — matrix operand order repaired and checked by reading, not compiled, not run on hardware
//
#include <hip/hip_runtime.h>
#include <hip/hip_bf16.h>

// ---------------------------------------------------------------------------
// 3D Swin block for MI455X (gfx1250, wave32, WMMA).
// All GEMMs run through V_WMMA_F32_16X16X32_BF16 (bf16 in, fp32 accumulate).
// GEMMs: 128x64 block tile, BK=64, double-buffered LDS (one barrier / stage,
// global->LDS stores for stage s+1 issued before the 8 WMMAs of stage s).
// Attention: flash-style, 2 blocks per (window, head), 256 threads = 8 waves
// x 32 q-rows. Full K and V^T tiles + P + rel-pos bias tables resident in
// 185 KB of LDS (CDNA5 WGP has 320 KB LDS -- impossible on CDNA4). Per-wave
// VGPR footprint ~200: no scratch spills, no s_set_vgpr_msb traffic.
// ---------------------------------------------------------------------------

typedef __bf16 bf16;
typedef __attribute__((ext_vector_type(16))) __bf16 v16bf;
typedef __attribute__((ext_vector_type(8)))  __bf16 v8bf;
typedef __attribute__((ext_vector_type(8)))  float  v8f;

#define DEV __device__ __forceinline__

DEV v8f wmma_bf16(v16bf a, v16bf b, v8f c) {
  return __builtin_amdgcn_wmma_f32_16x16x32_bf16(false, a, false, b,
                                                 (short)0, c, false, false);
}

// A-fragment (16x32, MxK), row-major source with leading dim ld (elements).
// ISA 7.12.2: lanes 0-15 row M=lane hold K 0..7 + 16..23; lanes 16-31 hold
// K 8..15 + 24..31.
DEV v16bf load_fragA(const bf16* p, int ld) {
  int lane = threadIdx.x & 31;
  const bf16* q = p + (lane & 15) * ld + ((lane >> 4) << 3);
  v8bf lo = *(const v8bf*)q;
  v8bf hi = *(const v8bf*)(q + 16);
  return __builtin_shufflevector(lo, hi, 0, 1, 2, 3, 4, 5, 6, 7,
                                 8, 9, 10, 11, 12, 13, 14, 15);
}

// B-fragment (32x16, KxN) loaded from an N-major ("Bt", N x K row-major)
// source: lane picks column N=lane&15, lanes 0-15 hold K 0..7 + 16..23, lanes
// 16-31 hold K 8..15 + 24..31 (matching the A-fragment K order).
DEV v16bf load_fragB(const bf16* p, int ld) {
  int lane = threadIdx.x & 31;
  const bf16* q = p + (lane & 15) * ld + ((lane >> 4) << 3);
  v8bf lo = *(const v8bf*)q;
  v8bf hi = *(const v8bf*)(q + 16);
  return __builtin_shufflevector(lo, hi, 0, 1, 2, 3, 4, 5, 6, 7,
                                 8, 9, 10, 11, 12, 13, 14, 15);
}

DEV int raw_to_winrow(int t) {
  int b = t >> 12, d = (t >> 8) & 15, h = (t >> 4) & 15, w = t & 15;
  int win = ((b * 2 + (d >> 3)) * 2 + (h >> 3)) * 2 + (w >> 3);
  int n = (((d & 7) * 8 + (h & 7)) * 8) + (w & 7);
  return win * 512 + n;
}
DEV int winrow_to_raw(int gr) {
  int b = gr >> 12, r = gr & 4095;
  int w3 = r >> 9, n = r & 511;
  int d = ((w3 >> 2) & 1) * 8 + (n >> 6);
  int h = ((w3 >> 1) & 1) * 8 + ((n >> 3) & 7);
  int w = (w3 & 1) * 8 + (n & 7);
  return ((b * 16 + d) * 16 + h) * 16 + w;
}

DEV float gelu_erf(float x) {
  return 0.5f * x * (1.0f + erff(x * 0.70710678118654752f));
}

// ---------------------------------------------------------------------------
// Weight convert + transpose: src (K x N fp32, row-major) -> dst (N x K bf16).
// ---------------------------------------------------------------------------
__global__ __launch_bounds__(256) void wt_bf16_kernel(
    const float* __restrict__ src, bf16* __restrict__ dst, int K, int N) {
  int idx = blockIdx.x * 256 + threadIdx.x;
  if (idx >= K * N) return;
  int n = idx / K, k = idx - n * K;
  dst[idx] = (bf16)src[(size_t)k * N + n];
}

// ---------------------------------------------------------------------------
// LayerNorm over C=768, one token per 256-thread block; output bf16.
// raw_to_win != 0: input rows are raw (b,d,h,w) order, output rows window
// order (LN1 fused with window partition). Otherwise rows pass through (LN2).
// ---------------------------------------------------------------------------
__global__ __launch_bounds__(256) void ln_kernel(
    const float* __restrict__ in, const float* __restrict__ w,
    const float* __restrict__ b, bf16* __restrict__ out, int raw_to_win) {
  __shared__ float red[8], red2[8];
  int t = blockIdx.x, tid = threadIdx.x;
  int lane = tid & 31, wid = tid >> 5;
  const float* row = in + (size_t)t * 768;
  float x0 = row[tid], x1 = row[tid + 256], x2 = row[tid + 512];
  float s = x0 + x1 + x2;
#pragma unroll
  for (int off = 16; off > 0; off >>= 1) s += __shfl_down(s, off);
  if (lane == 0) red[wid] = s;
  __syncthreads();
  float mean = 0.f;
#pragma unroll
  for (int i = 0; i < 8; ++i) mean += red[i];
  mean *= (1.f / 768.f);
  float d0 = x0 - mean, d1 = x1 - mean, d2 = x2 - mean;
  float ss = d0 * d0 + d1 * d1 + d2 * d2;
#pragma unroll
  for (int off = 16; off > 0; off >>= 1) ss += __shfl_down(ss, off);
  if (lane == 0) red2[wid] = ss;
  __syncthreads();
  float var = 0.f;
#pragma unroll
  for (int i = 0; i < 8; ++i) var += red2[i];
  var *= (1.f / 768.f);
  float inv = rsqrtf(var + 1e-5f);
  int orow = raw_to_win ? raw_to_winrow(t) : t;
  bf16* op = out + (size_t)orow * 768;
  op[tid]       = (bf16)(d0 * inv * w[tid]       + b[tid]);
  op[tid + 256] = (bf16)(d1 * inv * w[tid + 256] + b[tid + 256]);
  op[tid + 512] = (bf16)(d2 * inv * w[tid + 512] + b[tid + 512]);
}

// ---------------------------------------------------------------------------
// Generic bf16 WMMA GEMM:  C[M,N] = A[M,K] @ Bt[N,K]^T  (+ fused epilogue EP)
// Block tile 128x64, BK=64, double-buffered LDS, 256 threads = 8 waves in a
// 4x2 grid (32x32/wave, 8 v_wmma per stage per wave).
//  EP 0: QKV     -> +b_qkv, scatter to Q/K row-major and V^T per (win,head)
//  EP 1: PROJ    -> +b_proj + x (raw order), write resid1 fp32 (window order)
//  EP 2: MLP fc1 -> +b1, exact-erf GELU, write bf16
//  EP 3: MLP fc2 -> +b2 + resid1, un-window permute, write d_out fp32
// ---------------------------------------------------------------------------
template <int EP>
__global__ __launch_bounds__(256) void gemm_kernel(
    const bf16* __restrict__ A, const bf16* __restrict__ Bt, int K, int N,
    const float* __restrict__ bias, const float* __restrict__ add_src,
    float* __restrict__ out_f32, bf16* __restrict__ out_bf,
    bf16* __restrict__ qout, bf16* __restrict__ kout,
    bf16* __restrict__ vtout) {
  __shared__ bf16 As[2][128 * 72];  // 72-elem stride: bank-conflict-free
  __shared__ bf16 Bs[2][64 * 72];
  int tid = threadIdx.x;
  int lane = tid & 31, wid = tid >> 5;
  int wm = wid >> 1, wn = wid & 1;
  const bf16* Ab = A + (size_t)blockIdx.x * 128 * K;
  const bf16* Bb = Bt + (size_t)blockIdx.y * 64 * K;
  int arow = tid >> 1, acol = (tid & 1) * 32;  // 128x64: 32 elems/thread
  int brow = tid >> 2, bcol = (tid & 3) * 16;  // 64x64: 16 elems/thread
  v8f acc[2][2] = {};

  auto load_stage = [&](int buf, int k0) {
    const bf16* Ap = &Ab[(size_t)arow * K + k0 + acol];
    bf16* Ad = &As[buf][arow * 72 + acol];
    *(v8bf*)(Ad)      = *(const v8bf*)(Ap);
    *(v8bf*)(Ad + 8)  = *(const v8bf*)(Ap + 8);
    *(v8bf*)(Ad + 16) = *(const v8bf*)(Ap + 16);
    *(v8bf*)(Ad + 24) = *(const v8bf*)(Ap + 24);
    const bf16* Bp = &Bb[(size_t)brow * K + k0 + bcol];
    bf16* Bd = &Bs[buf][brow * 72 + bcol];
    *(v8bf*)(Bd)     = *(const v8bf*)(Bp);
    *(v8bf*)(Bd + 8) = *(const v8bf*)(Bp + 8);
  };

  load_stage(0, 0);
  __syncthreads();
  int nstage = K >> 6;
  for (int st = 0; st < nstage; ++st) {
    if (st + 1 < nstage) load_stage((st + 1) & 1, (st + 1) << 6);
    int buf = st & 1;
#pragma unroll
    for (int kk = 0; kk < 2; ++kk) {
      v16bf a0 = load_fragA(&As[buf][(wm * 32) * 72 + kk * 32], 72);
      v16bf a1 = load_fragA(&As[buf][(wm * 32 + 16) * 72 + kk * 32], 72);
      v16bf b0 = load_fragB(&Bs[buf][(wn * 32) * 72 + kk * 32], 72);
      v16bf b1 = load_fragB(&Bs[buf][(wn * 32 + 16) * 72 + kk * 32], 72);
      acc[0][0] = wmma_bf16(a0, b0, acc[0][0]);
      acc[0][1] = wmma_bf16(a0, b1, acc[0][1]);
      acc[1][0] = wmma_bf16(a1, b0, acc[1][0]);
      acc[1][1] = wmma_bf16(a1, b1, acc[1][1]);
    }
    __syncthreads();
  }

  int lc = lane & 15, r0 = (lane >> 4) << 3;
  int gr0 = blockIdx.x * 128 + wm * 32;
  int gc0 = blockIdx.y * 64 + wn * 32;
#pragma unroll
  for (int mi = 0; mi < 2; ++mi)
#pragma unroll
    for (int ni = 0; ni < 2; ++ni) {
      int gc = gc0 + ni * 16 + lc;
      float bv = bias[gc];
#pragma unroll
      for (int j = 0; j < 8; ++j) {
        int gr = gr0 + mi * 16 + r0 + j;
        float v = acc[mi][ni][j] + bv;
        if (EP == 0) {
          int t3 = gc / 768, rem = gc - t3 * 768;
          int head = rem >> 6, chn = rem & 63;
          int win = gr >> 9, n = gr & 511;
          size_t wh = (size_t)(win * 12 + head);
          if (t3 == 0)      qout[(wh * 512 + n) * 64 + chn] = (bf16)v;
          else if (t3 == 1) kout[(wh * 512 + n) * 64 + chn] = (bf16)v;
          else              vtout[(wh * 64 + chn) * 512 + n] = (bf16)v;
        } else if (EP == 1) {
          v += add_src[(size_t)winrow_to_raw(gr) * 768 + gc];
          out_f32[(size_t)gr * 768 + gc] = v;
        } else if (EP == 2) {
          out_bf[(size_t)gr * (size_t)N + gc] = (bf16)gelu_erf(v);
        } else {
          v += add_src[(size_t)gr * 768 + gc];
          out_f32[(size_t)winrow_to_raw(gr) * 768 + gc] = v;
        }
      }
    }
}

// ---------------------------------------------------------------------------
// Flash attention. Two blocks per (window, head): block handles 256 q-rows
// (half = blockIdx.x & 1), 256 threads = 8 waves x 32 q-rows each.
// Dynamic LDS (185,344 B):
//   Ks  [512][72] bf16    full K tile          73,728 B
//   Vts [64][520] bf16    full V^T tile        66,560 B
//   Ps  8x[32][40] bf16   per-wave P tiles     20,480 B
//   bD/bH/bW [256][8] f32 rel-pos bias dots    24,576 B
// ---------------------------------------------------------------------------
__global__ __launch_bounds__(256) void attn_kernel(
    const bf16* __restrict__ qg, const bf16* __restrict__ kg,
    const bf16* __restrict__ vtg, bf16* __restrict__ outg,
    const float* __restrict__ rpd, const float* __restrict__ rph,
    const float* __restrict__ rpw) {
  extern __shared__ char smem[];
  bf16* Ks  = (bf16*)smem;              // 512*72
  bf16* Vts = Ks + 512 * 72;            // 64*520
  bf16* Ps  = Vts + 64 * 520;           // 8*32*40
  float* bD = (float*)(Ps + 8 * 32 * 40);
  float* bH = bD + 256 * 8;
  float* bW = bH + 256 * 8;

  int tid = threadIdx.x, wid = tid >> 5, lane = tid & 31;
  int lc = lane & 15, r0 = (lane >> 4) << 3;
  int wh = blockIdx.x >> 1;
  int qrow0 = (blockIdx.x & 1) << 8;    // 0 or 256
  int win = wh / 12, head = wh - win * 12;
  const bf16* qb = qg + (size_t)wh * 512 * 64;
  const bf16* kbp = kg + (size_t)wh * 512 * 64;
  const bf16* vbp = vtg + (size_t)wh * 64 * 512;

  // Stage K and V^T into LDS (padded strides).
  for (int ch = tid; ch < 4096; ch += 256) {
    int r = ch >> 3, c8 = (ch & 7) * 8;
    *(v8bf*)&Ks[r * 72 + c8] = *(const v8bf*)&kbp[r * 64 + c8];
  }
  for (int ch = tid; ch < 4096; ch += 256) {
    int r = ch >> 6, c8 = (ch & 63) * 8;
    *(v8bf*)&Vts[r * 520 + c8] = *(const v8bf*)&vbp[r * 512 + c8];
  }
  // Decomposed rel-pos bias tables (this block's 256 q-rows only):
  // bX[nl][k] = sum_c q[qrow0+nl][c] * rel_X[i(n)-k+7][c]
  for (int e = tid; e < 256 * 24; e += 256) {
    int nl = e / 24, rem = e - nl * 24, dim = rem >> 3, k = rem & 7;
    int n = qrow0 + nl;
    int i = (dim == 0) ? (n >> 6) : (dim == 1) ? ((n >> 3) & 7) : (n & 7);
    const float* tab = (dim == 0) ? rpd : (dim == 1) ? rph : rpw;
    const float* tr = tab + (i - k + 7) * 64;
    const bf16* qr = qb + n * 64;
    float s = 0.f;
#pragma unroll 8
    for (int c = 0; c < 64; ++c) s += (float)qr[c] * tr[c];
    ((dim == 0) ? bD : (dim == 1) ? bH : bW)[nl * 8 + k] = s;
  }
  __syncthreads();

  // Per-wave Q fragments (32 rows x 64 ch) straight from global.
  int qbase = qrow0 + wid * 32;
  v16bf aq[2][2];
#pragma unroll
  for (int mi = 0; mi < 2; ++mi)
#pragma unroll
    for (int kk = 0; kk < 2; ++kk)
      aq[mi][kk] = load_fragA(qb + (size_t)(qbase + mi * 16) * 64 + kk * 32, 64);

  float m[2][8], l[2][8];
  v8f o[2][4] = {};
#pragma unroll
  for (int mi = 0; mi < 2; ++mi)
#pragma unroll
    for (int j = 0; j < 8; ++j) { m[mi][j] = -INFINITY; l[mi][j] = 0.f; }

  bf16* Pw = Ps + wid * 32 * 40;

  for (int kb = 0; kb < 16; ++kb) {
    // S = scale * Q K^T  (32 x 32 per wave)
    v8f s[2][2] = {};
#pragma unroll
    for (int nb = 0; nb < 2; ++nb)
#pragma unroll
      for (int kk = 0; kk < 2; ++kk) {
        v16bf bk = load_fragB(&Ks[(kb * 32 + nb * 16) * 72 + kk * 32], 72);
#pragma unroll
        for (int mi = 0; mi < 2; ++mi)
          s[mi][nb] = wmma_bf16(aq[mi][kk], bk, s[mi][nb]);
      }
    // scale + rel-pos bias (local-row indexed tables)
#pragma unroll
    for (int nb = 0; nb < 2; ++nb) {
      int kt = kb * 32 + nb * 16 + lc;
      int kd = kt >> 6, kh = (kt >> 3) & 7, kw = kt & 7;
#pragma unroll
      for (int mi = 0; mi < 2; ++mi)
#pragma unroll
        for (int j = 0; j < 8; ++j) {
          int nl = wid * 32 + mi * 16 + r0 + j;
          s[mi][nb][j] = s[mi][nb][j] * 0.125f +
                         bD[nl * 8 + kd] + bH[nl * 8 + kh] + bW[nl * 8 + kw];
        }
    }
    // Online softmax update; write P tile (bf16) to per-wave LDS region.
#pragma unroll
    for (int mi = 0; mi < 2; ++mi) {
#pragma unroll
      for (int j = 0; j < 8; ++j) {
        float mx = fmaxf(s[mi][0][j], s[mi][1][j]);
        mx = fmaxf(mx, __shfl_xor(mx, 1));
        mx = fmaxf(mx, __shfl_xor(mx, 2));
        mx = fmaxf(mx, __shfl_xor(mx, 4));
        mx = fmaxf(mx, __shfl_xor(mx, 8));
        float mn = fmaxf(m[mi][j], mx);
        float sc = __expf(m[mi][j] - mn);
        float p0 = __expf(s[mi][0][j] - mn);
        float p1 = __expf(s[mi][1][j] - mn);
        float rs = p0 + p1;
        rs += __shfl_xor(rs, 1);
        rs += __shfl_xor(rs, 2);
        rs += __shfl_xor(rs, 4);
        rs += __shfl_xor(rs, 8);
        l[mi][j] = l[mi][j] * sc + rs;
        m[mi][j] = mn;
#pragma unroll
        for (int ni = 0; ni < 4; ++ni) o[mi][ni][j] *= sc;
        int lrow = mi * 16 + r0 + j;
        Pw[lrow * 40 + lc]      = (bf16)p0;
        Pw[lrow * 40 + 16 + lc] = (bf16)p1;
      }
    }
    // O += P @ V  (K = 32 tokens of this block)
    v16bf bv[4];
#pragma unroll
    for (int ni = 0; ni < 4; ++ni)
      bv[ni] = load_fragB(&Vts[(ni * 16) * 520 + kb * 32], 520);
#pragma unroll
    for (int mi = 0; mi < 2; ++mi) {
      v16bf ap = load_fragA(&Pw[(mi * 16) * 40], 40);
#pragma unroll
      for (int ni = 0; ni < 4; ++ni)
        o[mi][ni] = wmma_bf16(ap, bv[ni], o[mi][ni]);
    }
  }

  // Normalize and write [win*512+n, head*64+ch] in bf16 (proj GEMM A input).
#pragma unroll
  for (int mi = 0; mi < 2; ++mi)
#pragma unroll
    for (int j = 0; j < 8; ++j) {
      float inv = 1.f / l[mi][j];
      int n = qbase + mi * 16 + r0 + j;
      size_t base = ((size_t)win * 512 + n) * 768 + head * 64;
#pragma unroll
      for (int ni = 0; ni < 4; ++ni)
        outg[base + ni * 16 + lc] = (bf16)(o[mi][ni][j] * inv);
    }
}

// ---------------------------------------------------------------------------
// Launch
// ---------------------------------------------------------------------------
extern "C" void kernel_launch(void* const* d_in, const int* in_sizes, int n_in,
                              void* d_out, int out_size, void* d_ws,
                              size_t ws_size, hipStream_t stream) {
  (void)in_sizes; (void)n_in; (void)out_size; (void)ws_size;
  const float* x      = (const float*)d_in[0];
  const float* w_qkv  = (const float*)d_in[1];
  const float* b_qkv  = (const float*)d_in[2];
  const float* w_proj = (const float*)d_in[3];
  const float* b_proj = (const float*)d_in[4];
  const float* rpd    = (const float*)d_in[5];
  const float* rph    = (const float*)d_in[6];
  const float* rpw    = (const float*)d_in[7];
  const float* n1w    = (const float*)d_in[8];
  const float* n1b    = (const float*)d_in[9];
  const float* n2w    = (const float*)d_in[10];
  const float* n2b    = (const float*)d_in[11];
  const float* w1     = (const float*)d_in[12];
  const float* b1     = (const float*)d_in[13];
  const float* w2     = (const float*)d_in[14];
  const float* b2     = (const float*)d_in[15];
  float* out = (float*)d_out;

  char* ws = (char*)d_ws;
  size_t off = 0;
  auto give = [&](size_t sz) { size_t r = off; off += (sz + 255) & ~(size_t)255; return r; };
  bf16* xw    = (bf16*)(ws + give(8192ull * 768 * 2));   // also reused as g
  bf16* qbuf  = (bf16*)(ws + give(192ull * 512 * 64 * 2));
  bf16* kbuf  = (bf16*)(ws + give(192ull * 512 * 64 * 2));
  bf16* vtbuf = (bf16*)(ws + give(192ull * 64 * 512 * 2));
  bf16* aout  = (bf16*)(ws + give(8192ull * 768 * 2));
  float* r1   = (float*)(ws + give(8192ull * 768 * 4));
  bf16* hbuf  = (bf16*)(ws + give(8192ull * 768 * 2));
  bf16* wqkvT = (bf16*)(ws + give(2304ull * 768 * 2));
  bf16* wprojT= (bf16*)(ws + give(768ull * 768 * 2));
  bf16* w1T   = (bf16*)(ws + give(3072ull * 768 * 2));
  bf16* w2T   = (bf16*)(ws + give(768ull * 3072 * 2));
  bf16* gbuf  = xw;  // 8192*3072 bf16 overlays xw+q+k+vT exactly (dead then)

  // 1. weights -> bf16, transposed to N x K
  wt_bf16_kernel<<<(768 * 2304 + 255) / 256, 256, 0, stream>>>(w_qkv, wqkvT, 768, 2304);
  wt_bf16_kernel<<<(768 * 768 + 255) / 256, 256, 0, stream>>>(w_proj, wprojT, 768, 768);
  wt_bf16_kernel<<<(768 * 3072 + 255) / 256, 256, 0, stream>>>(w1, w1T, 768, 3072);
  wt_bf16_kernel<<<(3072 * 768 + 255) / 256, 256, 0, stream>>>(w2, w2T, 3072, 768);

  // 2. LN1 fused with window partition
  ln_kernel<<<8192, 256, 0, stream>>>(x, n1w, n1b, xw, 1);

  // 3. QKV GEMM [8192x768]x[768x2304], scatter into Q/K/V^T per (win,head)
  gemm_kernel<0><<<dim3(64, 36), 256, 0, stream>>>(
      xw, wqkvT, 768, 2304, b_qkv, nullptr, nullptr, nullptr, qbuf, kbuf, vtbuf);

  // 4. Flash attention: 384 blocks (2 per (window,head)), 185,344 B dyn LDS
  attn_kernel<<<384, 256, 185344, stream>>>(qbuf, kbuf, vtbuf, aout, rpd, rph, rpw);

  // 5. proj + residual(x) -> resid1 (window order, fp32)
  gemm_kernel<1><<<dim3(64, 12), 256, 0, stream>>>(
      aout, wprojT, 768, 768, b_proj, x, r1, nullptr, nullptr, nullptr, nullptr);

  // 6. LN2
  ln_kernel<<<8192, 256, 0, stream>>>(r1, n2w, n2b, hbuf, 0);

  // 7. MLP fc1 + exact GELU
  gemm_kernel<2><<<dim3(64, 48), 256, 0, stream>>>(
      hbuf, w1T, 768, 3072, b1, nullptr, nullptr, gbuf, nullptr, nullptr, nullptr);

  // 8. MLP fc2 + residual + un-window permute -> d_out
  gemm_kernel<3><<<dim3(64, 12), 256, 0, stream>>>(
      gbuf, w2T, 3072, 768, b2, r1, out, nullptr, nullptr, nullptr, nullptr);
}
